// GCN_60902636257285
// MI455X (gfx1250) — matrix operand order repaired and checked by reading, not compiled, not run on hardware
//
#include <hip/hip_runtime.h>
#include <hip/hip_bf16.h>
#include <cstddef>

// ---------------------------------------------------------------------------
// GCN (2-layer) for MI455X / gfx1250.
//  - GEMMs use v_wmma_f32_16x16x32_bf16 with bf16 hi/lo split (~fp32 accuracy).
//  - Edge aggregation: L2-resident gather + global_atomic_add_f32 scatter.
// ---------------------------------------------------------------------------

typedef __attribute__((ext_vector_type(16))) __bf16 v16bf;
typedef __attribute__((ext_vector_type(8)))  float  v8f;

// ---------------- degree / normalization ----------------

__global__ void deg_init_kernel(float* __restrict__ deg, int n) {
    int i = blockIdx.x * blockDim.x + threadIdx.x;
    if (i < n) deg[i] = 1.0f;  // self-loop contributes 1 to every node's degree
}

__global__ void deg_count_kernel(const int* __restrict__ dst,
                                 float* __restrict__ deg, int E) {
    int e = blockIdx.x * blockDim.x + threadIdx.x;
    if (e < E) atomicAdd(&deg[dst[e]], 1.0f);
}

__global__ void deg_rsqrt_kernel(float* __restrict__ deg, int n) {
    int i = blockIdx.x * blockDim.x + threadIdx.x;
    if (i < n) deg[i] = rsqrtf(deg[i]);  // deg >= 1 always (self-loops)
}

// ---------------- WMMA GEMM: C[M,N] = A[M,K] @ B[K,N] ----------------
// One wave per 16x16 output tile; wave w of a block handles N-tile w.
// bf16 split: A ~= Ah + Al, B ~= Bh + Bl; acc += AhBh + AhBl + AlBh.
// EXACT_M=true removes all row clamps/guards (M % 16 == 0 -> straight-line
// loads/stores, EXEC stays all-ones, no per-store cmpx/branch).

template <int K_DIM, int N_DIM, bool EXACT_M>
__global__ void gemm_wmma_bf16split(const float* __restrict__ A,
                                    const float* __restrict__ B,
                                    float* __restrict__ C, int M) {
    const int lane = threadIdx.x & 31;
    const int wave = threadIdx.x >> 5;
    const int half = lane >> 4;   // 0: lanes 0-15, 1: lanes 16-31
    const int l16  = lane & 15;
    const int m0   = blockIdx.x * 16;
    const int n0   = wave * 16;

    // Clamp the A row (guarded variant only) so EXEC stays all-ones through
    // the WMMAs (ISA §7.12 requires EXEC == all 1s).
    const int mrow = EXACT_M ? (m0 + l16)
                             : ((m0 + l16 < M) ? (m0 + l16) : (M - 1));
    const float* arow = A + (size_t)mrow * K_DIM;

    v8f acc = {};

    #pragma unroll
    for (int kb = 0; kb < K_DIM; kb += 32) {
        v16bf ah, al, bh, bl;

        // A 16x32 bf16 layout (ISA §7.12.2): lane holds row M=l16;
        // VGPR v<4 -> K = half*8 + 2v(+1); v>=4 -> K = 16 + half*8 + 2(v-4)(+1).
        // Per lane this is two runs of 8 consecutive floats -> b128 loads.
        #pragma unroll
        for (int v = 0; v < 8; ++v) {
            const int k = (v < 4) ? (half * 8 + 2 * v)
                                  : (16 + half * 8 + 2 * (v - 4));
            const float x0 = arow[kb + k];
            const float x1 = arow[kb + k + 1];
            const __bf16 h0 = (__bf16)x0;
            const __bf16 h1 = (__bf16)x1;
            ah[2 * v]     = h0;
            ah[2 * v + 1] = h1;
            al[2 * v]     = (__bf16)(x0 - (float)h0);
            al[2 * v + 1] = (__bf16)(x1 - (float)h1);
        }

        // B 32x16 bf16 layout: lane holds column N=l16;
        // VGPR v<4 -> K = half*8 + 2v(+1); v>=4 -> K = 16 + half*8 + 2(v-4)(+1).
        #pragma unroll
        for (int v = 0; v < 8; ++v) {
            const int k = (v < 4) ? (half * 8 + 2 * v)
                                  : (16 + half * 8 + 2 * (v - 4));
            const float x0 = B[(size_t)(kb + k)     * N_DIM + n0 + l16];
            const float x1 = B[(size_t)(kb + k + 1) * N_DIM + n0 + l16];
            const __bf16 h0 = (__bf16)x0;
            const __bf16 h1 = (__bf16)x1;
            bh[2 * v]     = h0;
            bh[2 * v + 1] = h1;
            bl[2 * v]     = (__bf16)(x0 - (float)h0);
            bl[2 * v + 1] = (__bf16)(x1 - (float)h1);
        }

        acc = __builtin_amdgcn_wmma_f32_16x16x32_bf16(false, ah, false, bh,
                                                      (short)0, acc, false, false);
        acc = __builtin_amdgcn_wmma_f32_16x16x32_bf16(false, ah, false, bl,
                                                      (short)0, acc, false, false);
        acc = __builtin_amdgcn_wmma_f32_16x16x32_bf16(false, al, false, bh,
                                                      (short)0, acc, false, false);
    }

    // C/D 16x16 f32 layout: VGPR r, lanes 0-15 -> M=r, lanes 16-31 -> M=8+r; N=l16.
    #pragma unroll
    for (int r = 0; r < 8; ++r) {
        const int m = m0 + r + 8 * half;
        if (EXACT_M || m < M)
            C[(size_t)m * N_DIM + n0 + l16] = acc[r];
    }
}

// ---------------- self-loop + bias init:  out = dinv^2 * xw + bias ----------------

template <int D>
__global__ void self_loop_bias_kernel(const float* __restrict__ xw,
                                      const float* __restrict__ dinv,
                                      const float* __restrict__ bias,
                                      float* __restrict__ out, int n_nodes) {
    const int idx = blockIdx.x * blockDim.x + threadIdx.x;
    if (idx >= n_nodes * D) return;
    const int m = idx >> (D == 128 ? 7 : 6);
    const int d = idx & (D - 1);
    const float s = dinv[m];
    out[idx] = s * s * xw[idx] + bias[d];
}

// ---------------- edge scatter:  out[dst] += dinv[src]*dinv[dst] * xw[src] --------
// One wave per edge; each lane moves D/32 contiguous floats (vectorized gather,
// per-element fp32 L2 atomics for the scatter).

template <int D>
__global__ void edge_scatter_kernel(const float* __restrict__ xw,
                                    const int* __restrict__ src,
                                    const int* __restrict__ dst,
                                    const float* __restrict__ dinv,
                                    float* __restrict__ out, int E) {
    constexpr int PER = D / 32;
    const int e = blockIdx.x * (blockDim.x >> 5) + (threadIdx.x >> 5);
    if (e >= E) return;
    const int lane = threadIdx.x & 31;
    const int s = src[e];
    const int t = dst[e];
    const float norm = dinv[s] * dinv[t];

    const float* __restrict__ xr = xw + (size_t)s * D + lane * PER;
    float* __restrict__ orow     = out + (size_t)t * D + lane * PER;

    float v[PER];
    #pragma unroll
    for (int i = 0; i < PER; ++i) v[i] = xr[i];
    #pragma unroll
    for (int i = 0; i < PER; ++i) atomicAdd(orow + i, norm * v[i]);
}

// ---------------- ReLU ----------------

__global__ void relu_kernel(float* __restrict__ x, int n) {
    int i = blockIdx.x * blockDim.x + threadIdx.x;
    if (i < n) x[i] = fmaxf(x[i], 0.0f);
}

// ---------------- driver ----------------

extern "C" void kernel_launch(void* const* d_in, const int* in_sizes, int n_in,
                              void* d_out, int out_size, void* d_ws, size_t ws_size,
                              hipStream_t stream) {
    constexpr int IN_DIM  = 64;
    constexpr int HID_DIM = 128;
    constexpr int OUT_DIM = 64;

    const float* x  = (const float*)d_in[0];
    const int*   ei = (const int*)  d_in[1];
    const float* W1 = (const float*)d_in[2];
    const float* b1 = (const float*)d_in[3];
    const float* W2 = (const float*)d_in[4];
    const float* b2 = (const float*)d_in[5];
    float* out = (float*)d_out;

    const int N = in_sizes[0] / IN_DIM;   // 100000 nodes
    const int E = in_sizes[1] / 2;        // 800000 edges
    const int* src = ei;                  // edge_index[0]
    const int* dst = ei + E;              // edge_index[1]

    // Workspace carve-up (~103 MB): dinv | bufA (xw1, later xw2) | bufB (h)
    char* ws = (char*)d_ws;
    float* dinv = (float*)ws;
    size_t off  = (((size_t)N * sizeof(float)) + 255) & ~(size_t)255;
    float* bufA = (float*)(ws + off);
    off += (size_t)N * HID_DIM * sizeof(float);
    float* bufB = (float*)(ws + off);

    const int  T      = 256;
    const int  mtiles = (N + 15) / 16;
    const bool exact  = (N % 16) == 0;

    // 1) degrees -> dinv
    deg_init_kernel<<<(N + T - 1) / T, T, 0, stream>>>(dinv, N);
    deg_count_kernel<<<(E + T - 1) / T, T, 0, stream>>>(dst, dinv, E);
    deg_rsqrt_kernel<<<(N + T - 1) / T, T, 0, stream>>>(dinv, N);

    // 2) layer 1: xw1 = x @ W1   (8 N-tiles -> 8 waves/block)
    if (exact)
        gemm_wmma_bf16split<IN_DIM, HID_DIM, true><<<mtiles, 8 * 32, 0, stream>>>(x, W1, bufA, N);
    else
        gemm_wmma_bf16split<IN_DIM, HID_DIM, false><<<mtiles, 8 * 32, 0, stream>>>(x, W1, bufA, N);
    // h = dinv^2 * xw1 + b1  (self-loop + bias), then scatter edges, then ReLU
    self_loop_bias_kernel<HID_DIM><<<(N * HID_DIM + T - 1) / T, T, 0, stream>>>(
        bufA, dinv, b1, bufB, N);
    edge_scatter_kernel<HID_DIM><<<(E + 7) / 8, T, 0, stream>>>(
        bufA, src, dst, dinv, bufB, E);
    relu_kernel<<<(N * HID_DIM + T - 1) / T, T, 0, stream>>>(bufB, N * HID_DIM);

    // 3) layer 2: xw2 = h @ W2   (4 N-tiles -> 4 waves/block); reuse bufA
    if (exact)
        gemm_wmma_bf16split<HID_DIM, OUT_DIM, true><<<mtiles, 4 * 32, 0, stream>>>(bufB, W2, bufA, N);
    else
        gemm_wmma_bf16split<HID_DIM, OUT_DIM, false><<<mtiles, 4 * 32, 0, stream>>>(bufB, W2, bufA, N);
    // out = dinv^2 * xw2 + b2, then scatter edges directly into d_out
    self_loop_bias_kernel<OUT_DIM><<<(N * OUT_DIM + T - 1) / T, T, 0, stream>>>(
        bufA, dinv, b2, out, N);
    edge_scatter_kernel<OUT_DIM><<<(E + 7) / 8, T, 0, stream>>>(
        bufA, src, dst, dinv, out, E);
}
